// WordAttention_10531259810576
// MI455X (gfx1250) — matrix operand order repaired and checked by reading, not compiled, not run on hardware
//
#include <hip/hip_runtime.h>
#include <hip/hip_bf16.h>

typedef __bf16 bf16;
typedef __attribute__((ext_vector_type(16))) __bf16 v16bf;
typedef __attribute__((ext_vector_type(8)))  float  v8f;

#define BB   8
#define LP   4096
#define LQ   1024
#define DIN  300
#define DH   128
#define DPAD 304   // DIN padded to multiple of 16
#define NEG_INF (-__builtin_inff())

__device__ __forceinline__ v8f wmma_bf16(v16bf a, v16bf b, v8f c) {
    return __builtin_amdgcn_wmma_f32_16x16x32_bf16(
        false, a, false, b, (short)0, c, false, false);
}

// ---- async global->LDS (16B per active lane) via ISA mnemonic -------------
// vdst = 32-bit LDS byte offset (flat LDS addr low bits), vaddr = 64-bit VA.
__device__ __forceinline__ void cp16(void* l, const void* g) {
    unsigned lds_off      = (unsigned)(unsigned long long)l;
    unsigned long long ga = (unsigned long long)g;
    asm volatile("global_load_async_to_lds_b128 %0, %1, off"
                 :: "v"(lds_off), "v"(ga) : "memory");
}
__device__ __forceinline__ void async_wait0() {
    asm volatile("s_wait_asynccnt 0x0" ::: "memory");
}

// A fragment, 16x32 bf16, source row-major [m][k] with given stride (elements).
__device__ __forceinline__ v16bf load_A_frag(const bf16* base, int stride, int lane) {
    int m  = lane & 15;
    int kb = (lane >> 4) << 3;       // 0 or 8
    const bf16* r = base + (size_t)m * stride + kb;
    v16bf a;
#pragma unroll
    for (int j = 0; j < 8; ++j) a[j] = r[j];
#pragma unroll
    for (int j = 0; j < 8; ++j) a[8 + j] = r[16 + j];
    return a;
}

// B fragment, 32x16 bf16 (KxN). Source stored [n][k] row-major (k contiguous).
__device__ __forceinline__ v16bf load_B_frag(const bf16* base, int stride, int lane) {
    int n  = lane & 15;
    int kb = (lane >> 4) << 3;       // 0 or 8
    const bf16* r = base + (size_t)n * stride + kb;
    v16bf b;
#pragma unroll
    for (int j = 0; j < 8; ++j) b[j] = r[j];
#pragma unroll
    for (int j = 0; j < 8; ++j) b[8 + j] = r[16 + j];
    return b;
}

// ---------------- Projection: out = relu(x @ W^T + bias) as bf16 -------------
__global__ __launch_bounds__(128) void proj_kernel(
    const float* __restrict__ x, const float* __restrict__ W,
    const float* __restrict__ bias, bf16* __restrict__ out, int Lrows)
{
    const int ROWS = 64, KH = 160;
    int tiles = Lrows / ROWS;
    int batch = blockIdx.x / tiles;
    int row0  = (blockIdx.x % tiles) * ROWS;

    const float* xb = x + ((size_t)batch * Lrows + row0) * DIN;
    bf16*        ob = out + ((size_t)batch * Lrows + row0) * DH;

    __shared__ __align__(16) bf16 sW[DH][KH];    // 40 KB
    __shared__ __align__(16) bf16 sX[ROWS][KH];  // 20 KB

    int tid = threadIdx.x, lane = tid & 31, wave = tid >> 5;

    v8f acc[8];
#pragma unroll
    for (int t = 0; t < 8; ++t)
#pragma unroll
        for (int i = 0; i < 8; ++i) acc[t][i] = 0.f;

    for (int kh = 0; kh < 2; ++kh) {
        int kbase = kh * KH;
        for (int idx = tid; idx < DH * KH; idx += 128) {
            int n = idx / KH, k = idx % KH, kg = kbase + k;
            float v = (kg < DIN) ? W[(size_t)n * DIN + kg] : 0.f;
            sW[n][k] = (bf16)v;
        }
        for (int idx = tid; idx < ROWS * KH; idx += 128) {
            int m = idx / KH, k = idx % KH, kg = kbase + k;
            float v = (kg < DIN) ? xb[(size_t)m * DIN + kg] : 0.f;
            sX[m][k] = (bf16)v;
        }
        __syncthreads();

#pragma unroll
        for (int ks = 0; ks < 5; ++ks) {
            v16bf a = load_A_frag(&sX[wave * 16][0] + ks * 32, KH, lane);
            v16bf b = load_B_frag(&sW[0][0] + ks * 32, KH, lane);
#pragma unroll
            for (int t = 0; t < 8; ++t) {           // lookahead-1 pipeline
                v16bf nb = b;
                if (t < 7) nb = load_B_frag(&sW[(t + 1) * 16][0] + ks * 32, KH, lane);
                acc[t] = wmma_bf16(a, b, acc[t]);
                b = nb;
            }
        }
        __syncthreads();
    }

    int n = lane & 15;
    int mh = (lane >> 4) << 3;
#pragma unroll
    for (int t = 0; t < 8; ++t) {
        int col = t * 16 + n;
        float bv = bias[col];
#pragma unroll
        for (int i = 0; i < 8; ++i) {
            int m = wave * 16 + mh + i;
            float v = acc[t][i] + bv;
            ob[(size_t)m * DH + col] = (bf16)(v > 0.f ? v : 0.f);
        }
    }
}

// ------------- q -> bf16 transposed: qT[b][d][lq], d padded to 304 -----------
__global__ void qT_kernel(const float* __restrict__ q, bf16* __restrict__ qT)
{
    size_t idx = (size_t)blockIdx.x * blockDim.x + threadIdx.x;
    if (idx >= (size_t)BB * DPAD * LQ) return;
    int lq = (int)(idx % LQ);
    int d  = (int)((idx / LQ) % DPAD);
    int b  = (int)(idx / ((size_t)LQ * DPAD));
    float v = (d < DIN) ? q[((size_t)b * LQ + lq) * DIN + d] : 0.f;
    qT[idx] = (bf16)v;
}

// --------------- Flash attention: softmax(Wp·Wq^T + mask) · q ----------------
// 128 p-rows per workgroup (8 waves x 16 rows), q streamed in 32-row chunks,
// double-buffered async global->LDS staging.
__global__ __launch_bounds__(256) void attn_kernel(
    const bf16* __restrict__ Wp, const bf16* __restrict__ Wq,
    const bf16* __restrict__ qT, const unsigned char* __restrict__ q_mask,
    float* __restrict__ out)
{
    const int MT = 128, NT = DPAD / 16;  // 19 output tiles
    int tiles = LP / MT;
    int batch = blockIdx.x / tiles;
    int row0  = (blockIdx.x % tiles) * MT;

    __shared__ __align__(16) bf16  sWq[2][32][DH];    // 16 KB
    __shared__ __align__(16) bf16  sQT[2][DPAD][32];  // 38 KB
    __shared__ __align__(16) bf16  sP[8][16][32];     // 8 KB
    __shared__ float sMaskBias[2][32];

    int tid = threadIdx.x, lane = tid & 31, wave = tid >> 5;
    int n  = lane & 15;
    int mh = (lane >> 4) << 3;

    const bf16* WpB = Wp + ((size_t)batch * LP + row0) * DH;
    const bf16* WqB = Wq + (size_t)batch * LQ * DH;
    const bf16* qTB = qT + (size_t)batch * DPAD * LQ;
    const unsigned char* qmB = q_mask + (size_t)batch * LQ;

    auto issue_chunk = [&](int qb, int buf) {
        // Wq chunk: 32x128 bf16 contiguous = 512 x 16B
        for (int idx = tid; idx < 512; idx += 256)
            cp16(&sWq[buf][0][0] + idx * 8, WqB + (size_t)qb * DH + idx * 8);
        // q^T slab: 304 rows x 64B = 1216 x 16B
        for (int idx = tid; idx < 1216; idx += 256) {
            int d = idx >> 2, c = (idx & 3) * 8;
            cp16(&sQT[buf][d][c], qTB + (size_t)d * LQ + qb + c);
        }
        if (tid < 32) sMaskBias[buf][tid] = qmB[qb + tid] ? NEG_INF : 0.f;
    };

    // This wave's 16 rows of Wp as 4 A-fragments (K = 128), kept in registers.
    v16bf afrag[4];
#pragma unroll
    for (int ks = 0; ks < 4; ++ks)
        afrag[ks] = load_A_frag(WpB + (size_t)wave * 16 * DH + ks * 32, DH, lane);

    v8f o[NT];
#pragma unroll
    for (int t = 0; t < NT; ++t)
#pragma unroll
        for (int i = 0; i < 8; ++i) o[t][i] = 0.f;
    float rmax[8], rsum[8];
#pragma unroll
    for (int i = 0; i < 8; ++i) { rmax[i] = NEG_INF; rsum[i] = 0.f; }

    issue_chunk(0, 0);
    async_wait0();
    __syncthreads();

    for (int qb = 0; qb < LQ; qb += 32) {
        int cur = (qb >> 5) & 1, nxt = cur ^ 1;
        if (qb + 32 < LQ) issue_chunk(qb + 32, nxt);   // overlap with compute

        // S = Wp(16x128) @ Wq_chunk^T(128x32): two 16x16 C tiles, lookahead-1
        v8f s0, s1;
#pragma unroll
        for (int i = 0; i < 8; ++i) { s0[i] = 0.f; s1[i] = 0.f; }
        v16bf b0 = load_B_frag(&sWq[cur][0][0],  DH, lane);
        v16bf b1 = load_B_frag(&sWq[cur][16][0], DH, lane);
#pragma unroll
        for (int ks = 0; ks < 4; ++ks) {
            v16bf nb0 = b0, nb1 = b1;
            if (ks < 3) {
                nb0 = load_B_frag(&sWq[cur][0][0]  + (ks + 1) * 32, DH, lane);
                nb1 = load_B_frag(&sWq[cur][16][0] + (ks + 1) * 32, DH, lane);
            }
            s0 = wmma_bf16(afrag[ks], b0, s0);
            s1 = wmma_bf16(afrag[ks], b1, s1);
            b0 = nb0; b1 = nb1;
        }

        // masked online softmax
        float mb0 = sMaskBias[cur][n], mb1 = sMaskBias[cur][16 + n];
        float sc[8], nmexp[8];
#pragma unroll
        for (int i = 0; i < 8; ++i) {
            s0[i] += mb0; s1[i] += mb1;
            float m = fmaxf(s0[i], s1[i]);
#pragma unroll
            for (int off = 1; off < 16; off <<= 1)
                m = fmaxf(m, __shfl_xor(m, off, 32));
            float nm = fmaxf(rmax[i], m);
            nmexp[i] = (nm == NEG_INF) ? 0.f : nm;
            sc[i] = (rmax[i] == NEG_INF) ? 0.f : __expf(rmax[i] - nmexp[i]);
            rmax[i] = nm;
            rsum[i] *= sc[i];
        }
#pragma unroll
        for (int t = 0; t < NT; ++t)
#pragma unroll
            for (int i = 0; i < 8; ++i) o[t][i] *= sc[i];

        // P = exp(S - max) -> LDS (C layout -> A layout), accumulate row sums
#pragma unroll
        for (int i = 0; i < 8; ++i) {
            float p0 = __expf(s0[i] - nmexp[i]);
            float p1 = __expf(s1[i] - nmexp[i]);
            sP[wave][mh + i][n]      = (bf16)p0;
            sP[wave][mh + i][16 + n] = (bf16)p1;
            float r = p0 + p1;
#pragma unroll
            for (int off = 1; off < 16; off <<= 1)
                r += __shfl_xor(r, off, 32);
            rsum[i] += r;
        }

        // O(16x304) += P(16x32) @ q_chunk(32x304), lookahead-1
        v16bf pa = load_A_frag(&sP[wave][0][0], 32, lane);
        v16bf qf = load_B_frag(&sQT[cur][0][0], 32, lane);
#pragma unroll
        for (int t = 0; t < NT; ++t) {
            v16bf nqf = qf;
            if (t < NT - 1) nqf = load_B_frag(&sQT[cur][(t + 1) * 16][0], 32, lane);
            o[t] = wmma_bf16(pa, qf, o[t]);
            qf = nqf;
        }

        async_wait0();      // next chunk's async copies landed
        __syncthreads();    // everyone done reading 'cur' / writing 'nxt'
    }

    // normalize and store fp32 output
    float inv[8];
#pragma unroll
    for (int i = 0; i < 8; ++i) inv[i] = (rsum[i] > 0.f) ? 1.f / rsum[i] : 0.f;
    float* outB = out + ((size_t)batch * LP + row0) * DIN;
    int mbase = wave * 16 + mh;
#pragma unroll
    for (int t = 0; t < NT; ++t) {
        int d = t * 16 + n;
        if (d < DIN) {
#pragma unroll
            for (int i = 0; i < 8; ++i)
                outB[(size_t)(mbase + i) * DIN + d] = o[t][i] * inv[i];
        }
    }
}

extern "C" void kernel_launch(void* const* d_in, const int* in_sizes, int n_in,
                              void* d_out, int out_size, void* d_ws, size_t ws_size,
                              hipStream_t stream) {
    const float* p          = (const float*)d_in[0];
    // d_in[1] = p_mask (unused by the reference)
    const float* q          = (const float*)d_in[2];
    const unsigned char* qm = (const unsigned char*)d_in[3];
    const float* W          = (const float*)d_in[4];
    const float* bias       = (const float*)d_in[5];
    float* out              = (float*)d_out;

    char* ws = (char*)d_ws;
    bf16* Wp = (bf16*)ws;                                   // 8 MB
    bf16* Wq = (bf16*)(ws + (size_t)BB * LP * DH * 2);      // 2 MB
    bf16* qT = (bf16*)(ws + (size_t)BB * LP * DH * 2
                          + (size_t)BB * LQ * DH * 2);      // ~5 MB

    proj_kernel<<<BB * (LP / 64), 128, 0, stream>>>(p, W, bias, Wp, LP);
    proj_kernel<<<BB * (LQ / 64), 128, 0, stream>>>(q, W, bias, Wq, LQ);
    qT_kernel<<<(BB * DPAD * LQ + 255) / 256, 256, 0, stream>>>(q, qT);
    attn_kernel<<<BB * (LP / 128), 256, 0, stream>>>(Wp, Wq, qT, qm, out);
}
